// MultiHeadSelfAttention_90460601188918
// MI455X (gfx1250) — compile-verified, matrix operand order checked by reading, not run on hardware
//
#include <hip/hip_runtime.h>
#include <hip/hip_bf16.h>

// MI455X / gfx1250 fused multi-head self-attention.
// B=4, S=2048, D=1024, H=16, DK=64. bf16 WMMA (16x16x32), f32 accum.
// Attention K/V tiles staged LDS-side via the Tensor Data Mover (TDM).

typedef __bf16 bf16;
typedef __attribute__((ext_vector_type(16))) __bf16 v16bf;
typedef __attribute__((ext_vector_type(8)))  float  v8f;
typedef __attribute__((ext_vector_type(4)))  unsigned int u32x4;
typedef __attribute__((ext_vector_type(8)))  int i32x8;
typedef __attribute__((ext_vector_type(4)))  int i32x4;

#define S_LEN  2048
#define DMODEL 1024
#define NHEAD  16
#define DKH    64
#define BATCH  4
#define NROWS  (BATCH * S_LEN) /* 8192 */

union ABFrag { v16bf v; unsigned int u[8]; };

// Load a 16x32 bf16 fragment in the CDNA5 A-layout from a row-major matrix.
// (Also used for B-fragments when the source matrix is stored n-major, i.e.
//  pre-transposed, since B mirrors A's per-lane K-pair striping.)
__device__ __forceinline__ ABFrag load_frag_rm(const bf16* __restrict__ base,
                                               int ld, int row0, int k0) {
  const int lane = threadIdx.x & 31;
  const int m  = lane & 15;
  const int kb = (lane >> 4) << 3;            // lanes 16-31 cover K+8 / K+24
  const bf16* p = base + (size_t)(row0 + m) * ld + k0 + kb;
  ABFrag f;
#pragma unroll
  for (int v = 0; v < 8; ++v) {
    const int kk = (v < 4) ? (v << 1) : (16 + ((v - 4) << 1));
    f.u[v] = *reinterpret_cast<const unsigned int*>(p + kk);   // 2 bf16 / VGPR
  }
  return f;
}

__device__ __forceinline__ v8f wmma_bf16(const ABFrag& a, const ABFrag& b, v8f c) {
  return __builtin_amdgcn_wmma_f32_16x16x32_bf16(
      /*neg_a=*/false, a.v, /*neg_b=*/false, b.v,
      /*c_mod=*/(short)0, c, /*reuse_a=*/false, /*reuse_b=*/false);
}

// ---- Tensor Data Mover: 2-D tile load Global -> LDS (D# per ISA 08, §8) ----
// lds_off      : byte offset of tile start within the workgroup LDS block
// gptr         : global address of tile start (bf16 elements)
// tile_w/h     : tile width (elements, contiguous) / height (rows)
// row_stride   : tensor dim0 stride in elements
// tensor_h     : tensor dim1 extent (rows) for OOB clamping
__device__ __forceinline__ void tdm_load_2d(unsigned lds_off, const bf16* gptr,
                                            unsigned tile_w, unsigned tile_h,
                                            unsigned long long row_stride,
                                            unsigned tensor_h) {
  const unsigned long long ga = (unsigned long long)(size_t)gptr;
  const unsigned td0 = (unsigned)row_stride;       // tensor_dim0 (elements)
  const unsigned td1 = tensor_h;                   // tensor_dim1 (rows)
  // Group 0: count=1 | lds_addr | global_addr[56:0] | type=2 ("image")
  unsigned g0w0 = 1u;
  unsigned g0w1 = lds_off;
  unsigned g0w2 = (unsigned)(ga & 0xFFFFFFFFu);
  unsigned g0w3 = (unsigned)((ga >> 32) & 0x1FFFFFFu) | (2u << 30);
  // Group 1: wg_mask=0 | data_size=1 (2 bytes) | no pad/barrier | dims/strides
  unsigned g1w0 = (1u << 16);                                    // data_size=2B
  unsigned g1w1 = (td0 & 0xFFFFu) << 16;                         // td0[15:0]
  unsigned g1w2 = (td0 >> 16) | ((td1 & 0xFFFFu) << 16);         // td0[31:16]|td1[15:0]
  unsigned g1w3 = (td1 >> 16) | (tile_w << 16);                  // td1[31:16]|tile_dim0
  unsigned g1w4 = tile_h & 0xFFFFu;                              // tile_dim1, tile_dim2=0
  unsigned g1w5 = (unsigned)(row_stride & 0xFFFFFFFFull);        // dim0_stride[31:0]
  unsigned g1w6 = (unsigned)(row_stride >> 32);                  // dim0_stride[47:32]
  unsigned g1w7 = 0;                                             // dim1_stride unused (2D)
  // Descriptors live in SGPRs: force wave-uniform values.
  u32x4 g0 = { (unsigned)__builtin_amdgcn_readfirstlane((int)g0w0),
               (unsigned)__builtin_amdgcn_readfirstlane((int)g0w1),
               (unsigned)__builtin_amdgcn_readfirstlane((int)g0w2),
               (unsigned)__builtin_amdgcn_readfirstlane((int)g0w3) };
  i32x8 g1 = { __builtin_amdgcn_readfirstlane((int)g1w0),
               __builtin_amdgcn_readfirstlane((int)g1w1),
               __builtin_amdgcn_readfirstlane((int)g1w2),
               __builtin_amdgcn_readfirstlane((int)g1w3),
               __builtin_amdgcn_readfirstlane((int)g1w4),
               __builtin_amdgcn_readfirstlane((int)g1w5),
               __builtin_amdgcn_readfirstlane((int)g1w6),
               __builtin_amdgcn_readfirstlane((int)g1w7) };
  i32x4 z4 = { 0, 0, 0, 0 };                       // groups 2/3 unused (2-D tile)
  i32x8 z8 = { 0, 0, 0, 0, 0, 0, 0, 0 };           // 6-arg toolchain extra group
  __builtin_amdgcn_tensor_load_to_lds(g0, g1, z4, z4, z8, 0);
}

// ---------------------------------------------------------------- pack kernels
__global__ __launch_bounds__(256) void pack_x_kernel(const float* __restrict__ x,
                                                     bf16* __restrict__ xb, int n) {
  int i = blockIdx.x * 256 + threadIdx.x;
  if (i < n) xb[i] = (bf16)x[i];
}

// Wt[n*1024 + k] = bf16( W[k*1024 + n] )  (n-major for B-fragment loads)
__global__ __launch_bounds__(256) void pack_wT_kernel(const float* __restrict__ W,
                                                      bf16* __restrict__ Wt) {
  int o = blockIdx.x * 256 + threadIdx.x;   // 1024*1024 threads
  int nn = o >> 10, kk = o & 1023;
  Wt[o] = (bf16)W[kk * DMODEL + nn];
}

// ------------------------------------------------------------- QKV projection
// One wave computes a 32x64 tile (2 A-frags x 4 B-frags -> 8 WMMA / k-step),
// writing head-major layouts: Q,K -> [b,h,s,dk]; V -> [b,h,dk,s] (transposed).
__global__ __launch_bounds__(128) void qkv_kernel(
    const bf16* __restrict__ xb,
    const bf16* __restrict__ WqT, const bf16* __restrict__ WkT,
    const bf16* __restrict__ WvT,
    const float* __restrict__ bq, const float* __restrict__ bk,
    const float* __restrict__ bv,
    bf16* __restrict__ Q, bf16* __restrict__ K, bf16* __restrict__ Vt) {
  const int lane = threadIdx.x & 31;
  const int r0 = blockIdx.x * 32;
  const int c0 = (blockIdx.y * 4 + threadIdx.y) * 64;
  const int z  = blockIdx.z;
  const bf16*  Wt   = (z == 0) ? WqT : (z == 1) ? WkT : WvT;
  const float* bias = (z == 0) ? bq  : (z == 1) ? bk  : bv;

  v8f acc[2][4] = {};
  for (int kt = 0; kt < DMODEL; kt += 32) {
    ABFrag a0 = load_frag_rm(xb, DMODEL, r0,      kt);
    ABFrag a1 = load_frag_rm(xb, DMODEL, r0 + 16, kt);
#pragma unroll
    for (int t = 0; t < 4; ++t) {
      ABFrag b = load_frag_rm(Wt, DMODEL, c0 + t * 16, kt);  // reused twice
      acc[0][t] = wmma_bf16(a0, b, acc[0][t]);
      acc[1][t] = wmma_bf16(a1, b, acc[1][t]);
    }
  }

  const int hi = lane >> 4, n = lane & 15;
#pragma unroll
  for (int rr = 0; rr < 2; ++rr) {
#pragma unroll
    for (int t = 0; t < 4; ++t) {
      const int c  = c0 + t * 16 + n;
      const float bb = bias[c];
      const int h = c >> 6, dk = c & 63;
#pragma unroll
      for (int v = 0; v < 8; ++v) {
        const int row = r0 + rr * 16 + v + (hi << 3);
        const int b_ = row >> 11, s = row & (S_LEN - 1);
        const int bh = b_ * NHEAD + h;
        const bf16 val = (bf16)(acc[rr][t][v] + bb);
        if (z == 0)      Q [((size_t)bh * S_LEN + s) * DKH + dk] = val;
        else if (z == 1) K [((size_t)bh * S_LEN + s) * DKH + dk] = val;
        else             Vt[((size_t)bh * DKH + dk) * S_LEN + s] = val;
      }
    }
  }
}

// ---------------------------------------------------------- flash attention
// One wave owns a 16-query tile of one (b,h); iterates causal KV blocks of 32.
// K (32x64) and V^T (64x32) tiles are DMA'd into wave-private LDS by the TDM.
// Per-wave LDS region (elements): [0,2048) K tile, [2048,4096) V^T tile,
// [4096,4608) P staging. 4 waves * 9216 B = 36 KB.
#define WAVE_LDS_ELEMS 4608
__global__ __launch_bounds__(128) void attn_kernel(
    const bf16* __restrict__ Q, const bf16* __restrict__ K,
    const bf16* __restrict__ Vt, bf16* __restrict__ attn) {
  __shared__ __align__(16) bf16 smem[4 * WAVE_LDS_ELEMS];
  const int lane = threadIdx.x & 31;
  const int w = threadIdx.y;
  const int bh = blockIdx.y;
  const int qbase = (blockIdx.x * 4 + w) * 16;
  const bf16* Qb = Q  + (size_t)bh * S_LEN * DKH;
  const bf16* Kb = K  + (size_t)bh * S_LEN * DKH;
  const bf16* Vb = Vt + (size_t)bh * DKH * S_LEN;

  bf16* Ktile = smem + w * WAVE_LDS_ELEMS;          // 32 x 64
  bf16* Vtile = Ktile + 2048;                       // 64 x 32
  bf16* pbuf  = Ktile + 4096;                       // 16 x 32
  const unsigned ldsK = (unsigned)(w * WAVE_LDS_ELEMS) * 2u;        // byte offs
  const unsigned ldsV = ldsK + 4096u;

  ABFrag q0 = load_frag_rm(Qb, DKH, qbase, 0);
  ABFrag q1 = load_frag_rm(Qb, DKH, qbase, 32);

  float mI[8], lI[8];
#pragma unroll
  for (int v = 0; v < 8; ++v) { mI[v] = -1e30f; lI[v] = 0.f; }
  v8f acc[4] = {};

  const int hi = lane >> 4, n = lane & 15;
  const int jend = qbase + 16;
  for (int jb = 0; jb < jend; jb += 32) {
    // TDM: stage K rows [jb, jb+32) and V^T cols [jb, jb+32) into LDS.
    tdm_load_2d(ldsK, Kb + (size_t)jb * DKH, DKH, 32, DKH, S_LEN - jb);
    tdm_load_2d(ldsV, Vb + jb, 32, DKH, S_LEN, DKH);
    __builtin_amdgcn_s_wait_tensorcnt((short)0);
    asm volatile("" ::: "memory");

    // scores = Q(16x64) x K^T(64x32): K rows read from LDS as B-fragments
    ABFrag b00 = load_frag_rm(Ktile, DKH, 0,  0);
    ABFrag b01 = load_frag_rm(Ktile, DKH, 0,  32);
    ABFrag b10 = load_frag_rm(Ktile, DKH, 16, 0);
    ABFrag b11 = load_frag_rm(Ktile, DKH, 16, 32);
    v8f zc = {};
    v8f s0 = wmma_bf16(q0, b00, wmma_bf16(q1, b01, zc));  // kv n = 0..15
    v8f s1 = wmma_bf16(q0, b10, wmma_bf16(q1, b11, zc));  // kv n = 16..31

    float mNew[8], alpha[8], p0[8], p1[8];
#pragma unroll
    for (int v = 0; v < 8; ++v) {
      const int qi = qbase + v + (hi << 3);
      float a = s0[v] * 0.125f;                 // 1/sqrt(64)
      float b = s1[v] * 0.125f;
      if (jb + n > qi)      a = -1e9f;          // causal mask
      if (jb + 16 + n > qi) b = -1e9f;
      s0[v] = a; s1[v] = b;
      float mx = fmaxf(a, b);                   // row max across 16 lanes
      mx = fmaxf(mx, __shfl_xor(mx, 1, 32));
      mx = fmaxf(mx, __shfl_xor(mx, 2, 32));
      mx = fmaxf(mx, __shfl_xor(mx, 4, 32));
      mx = fmaxf(mx, __shfl_xor(mx, 8, 32));
      mNew[v]  = fmaxf(mI[v], mx);
      alpha[v] = __expf(mI[v] - mNew[v]);
      p0[v] = __expf(s0[v] - mNew[v]);
      p1[v] = __expf(s1[v] - mNew[v]);
      float sum = p0[v] + p1[v];
      sum += __shfl_xor(sum, 1, 32);
      sum += __shfl_xor(sum, 2, 32);
      sum += __shfl_xor(sum, 4, 32);
      sum += __shfl_xor(sum, 8, 32);
      lI[v] = lI[v] * alpha[v] + sum;
      mI[v] = mNew[v];
    }

    // Stage P in LDS (C-layout writes), reload in A-layout. Wave-private.
#pragma unroll
    for (int v = 0; v < 8; ++v) {
      const int row = v + (hi << 3);
      pbuf[row * 32 + n]      = (bf16)p0[v];
      pbuf[row * 32 + 16 + n] = (bf16)p1[v];
    }
    asm volatile("s_wait_dscnt 0" ::: "memory");  // CDNA5 split DS counter
    ABFrag pA;
    {
      const int m = lane & 15, kb = hi << 3;
#pragma unroll
      for (int v = 0; v < 8; ++v) {
        const int kk = ((v < 4) ? (v << 1) : (16 + ((v - 4) << 1))) + kb;
        pA.u[v] = *reinterpret_cast<const unsigned int*>(&pbuf[m * 32 + kk]);
      }
    }

    // O += P(16x32) @ V(32x64); V^T tile in LDS is n-major -> B-fragments
#pragma unroll
    for (int t = 0; t < 4; ++t) {
      ABFrag vb = load_frag_rm(Vtile, 32, t * 16, 0);
#pragma unroll
      for (int v = 0; v < 8; ++v) acc[t][v] *= alpha[v];
      acc[t] = wmma_bf16(pA, vb, acc[t]);
    }
  }

  // Write combined-head layout [b*S + s][h*64 + dk] (bf16) for output GEMM.
  const int b_ = bh >> 4, h = bh & 15;
#pragma unroll
  for (int t = 0; t < 4; ++t) {
#pragma unroll
    for (int v = 0; v < 8; ++v) {
      const int srow = qbase + v + (hi << 3);
      const int col  = h * DKH + t * 16 + n;
      attn[((size_t)b_ * S_LEN + srow) * DMODEL + col] = (bf16)(acc[t][v] / lI[v]);
    }
  }
}

// ---------------------------------------------------------- output projection
__global__ __launch_bounds__(128) void oproj_kernel(
    const bf16* __restrict__ attn, const bf16* __restrict__ WoT,
    const float* __restrict__ bo, float* __restrict__ out) {
  const int lane = threadIdx.x & 31;
  const int r0 = blockIdx.x * 32;
  const int c0 = (blockIdx.y * 4 + threadIdx.y) * 64;
  v8f acc[2][4] = {};
  for (int kt = 0; kt < DMODEL; kt += 32) {
    ABFrag a0 = load_frag_rm(attn, DMODEL, r0,      kt);
    ABFrag a1 = load_frag_rm(attn, DMODEL, r0 + 16, kt);
#pragma unroll
    for (int t = 0; t < 4; ++t) {
      ABFrag b = load_frag_rm(WoT, DMODEL, c0 + t * 16, kt);
      acc[0][t] = wmma_bf16(a0, b, acc[0][t]);
      acc[1][t] = wmma_bf16(a1, b, acc[1][t]);
    }
  }
  const int hi = lane >> 4, n = lane & 15;
#pragma unroll
  for (int rr = 0; rr < 2; ++rr) {
#pragma unroll
    for (int t = 0; t < 4; ++t) {
      const int c = c0 + t * 16 + n;
      const float bb = bo[c];
#pragma unroll
      for (int v = 0; v < 8; ++v) {
        const int row = r0 + rr * 16 + v + (hi << 3);
        out[(size_t)row * DMODEL + c] = acc[rr][t][v] + bb;
      }
    }
  }
}

// --------------------------------------------------------------------- launch
extern "C" void kernel_launch(void* const* d_in, const int* in_sizes, int n_in,
                              void* d_out, int out_size, void* d_ws, size_t ws_size,
                              hipStream_t stream) {
  (void)in_sizes; (void)n_in; (void)out_size; (void)ws_size;
  const float* x  = (const float*)d_in[0];
  // d_in[1] = attn_mask (causal; encoded directly in the kernel)
  const float* Wq = (const float*)d_in[2];
  const float* bq = (const float*)d_in[3];
  const float* Wk = (const float*)d_in[4];
  const float* bk = (const float*)d_in[5];
  const float* Wv = (const float*)d_in[6];
  const float* bv = (const float*)d_in[7];
  const float* Wo = (const float*)d_in[8];
  const float* bo = (const float*)d_in[9];

  char* ws = (char*)d_ws;
  size_t off = 0;
  bf16* xb   = (bf16*)(ws + off); off += (size_t)NROWS * DMODEL * 2;   // 16 MB
  bf16* WqT  = (bf16*)(ws + off); off += (size_t)DMODEL * DMODEL * 2;  //  2 MB
  bf16* WkT  = (bf16*)(ws + off); off += (size_t)DMODEL * DMODEL * 2;
  bf16* WvT  = (bf16*)(ws + off); off += (size_t)DMODEL * DMODEL * 2;
  bf16* WoT  = (bf16*)(ws + off); off += (size_t)DMODEL * DMODEL * 2;
  bf16* Qh   = (bf16*)(ws + off); off += (size_t)NROWS * DMODEL * 2;   // 16 MB
  bf16* Kh   = (bf16*)(ws + off); off += (size_t)NROWS * DMODEL * 2;
  bf16* Vth  = (bf16*)(ws + off); off += (size_t)NROWS * DMODEL * 2;
  bf16* attn = (bf16*)(ws + off); off += (size_t)NROWS * DMODEL * 2;

  const int nX = NROWS * DMODEL;
  pack_x_kernel<<<nX / 256, 256, 0, stream>>>(x, xb, nX);
  const int nW = DMODEL * DMODEL;
  pack_wT_kernel<<<nW / 256, 256, 0, stream>>>(Wq, WqT);
  pack_wT_kernel<<<nW / 256, 256, 0, stream>>>(Wk, WkT);
  pack_wT_kernel<<<nW / 256, 256, 0, stream>>>(Wv, WvT);
  pack_wT_kernel<<<nW / 256, 256, 0, stream>>>(Wo, WoT);

  qkv_kernel<<<dim3(NROWS / 32, 4, 3), dim3(32, 4), 0, stream>>>(
      xb, WqT, WkT, WvT, bq, bk, bv, Qh, Kh, Vth);

  attn_kernel<<<dim3(S_LEN / 64, BATCH * NHEAD), dim3(32, 4), 0, stream>>>(
      Qh, Kh, Vth, attn);

  oproj_kernel<<<dim3(NROWS / 32, 4), dim3(32, 4), 0, stream>>>(
      attn, WoT, bo, (float*)d_out);
}
